// Model_68049461837947
// MI455X (gfx1250) — compile-verified, matrix operand order checked by reading, not run on hardware
//
#include <hip/hip_runtime.h>
#include <math.h>

// ---------------------------------------------------------------------------
// Informer encoder forward for MI455X (gfx1250, wave32, WMMA).
// Big GEMMs -> v_wmma_f32_16x16x32_bf16, bf16 operands pre-converted ONCE
// (kills per-tile f32->bf16 VALU and halves LDS staging traffic).
// Wave tile 32x64 = 2x4 accumulators -> 8 WMMA per wave per K-step.
// Attention core (ProbSparse, u=35 queries) is tiny -> scalar kernels.
// Head GEMM (28x96x262144) is HBM-bound on Wh -> split-K with atomics.
// ---------------------------------------------------------------------------

#define BSZc   4
#define SEQc   512
#define NVARSc 7
#define DMc    512
#define NHc    8
#define DFFc   2048
#define ELc    2
#define PREDc  96
#define DHc    64
#define BVc    (BSZc * NVARSc)   // 28
#define MROWSc (BVc * SEQc)      // 14336
#define USELc  35                // u == U_part == 5*ceil(ln(512)) == 35
#define KTOTc  (SEQc * DMc)      // 262144

typedef __attribute__((ext_vector_type(16))) __bf16       bf16x16;
typedef __attribute__((ext_vector_type(8)))  float        f32x8;
typedef __attribute__((ext_vector_type(4)))  float        f32x4;
typedef __attribute__((ext_vector_type(4)))  unsigned int u32x4;
typedef __attribute__((ext_vector_type(2)))  unsigned int u32x2;

__device__ __forceinline__ unsigned short f2bf(float f) {
    unsigned u = __float_as_uint(f);
    unsigned r = u + 0x7FFFu + ((u >> 16) & 1u);   // round-to-nearest-even
    return (unsigned short)(r >> 16);
}

__device__ __forceinline__ float gelu_tanh(float x) {
    const float c = 0.7978845608028654f;           // sqrt(2/pi)
    float x3 = x * x * x;
    return 0.5f * x * (1.0f + tanhf(c * (x + 0.044715f * x3)));
}

__device__ __forceinline__ unsigned hash_u32(unsigned x) {
    x ^= x >> 16; x *= 0x7feb352du;
    x ^= x >> 15; x *= 0x846ca68bu;
    x ^= x >> 16;
    return x;
}

// fp32 -> bf16 bulk convert (n % 4 == 0 at all call sites)
__global__ __launch_bounds__(256) void convert_bf16_kernel(
    const float* __restrict__ x, unsigned short* __restrict__ y, int n)
{
    const int i = (blockIdx.x * 256 + threadIdx.x) * 4;
    if (i >= n) return;
    const f32x4 v = *(const f32x4*)(x + i);
    u32x2 p;
    p.x = (unsigned)f2bf(v.x) | ((unsigned)f2bf(v.y) << 16);
    p.y = (unsigned)f2bf(v.z) | ((unsigned)f2bf(v.w) << 16);
    *(u32x2*)(y + i) = p;
}

// ---------------------------------------------------------------------------
// WMMA GEMM: C[M,N] = act(A[M,K] @ W[N,K]^T + bias[N])
// A,W bf16 row-major (pre-converted). M%64==0, N%128==0, K%32==0.
// Block = 128 threads = 4 waves; block tile 64x128; K-step 32.
// Wave tile 32x64 = 2x4 v_wmma_f32_16x16x32_bf16 accumulators.
// LDS rows padded to 40 ushort (80B) -> conflict-free b128 fragment loads.
// Output: fp32 C (if non-null) and/or bf16 Cbf (if non-null).
// ---------------------------------------------------------------------------
__global__ __launch_bounds__(128) void gemm_bf16_wmma(
    const unsigned short* __restrict__ A, const unsigned short* __restrict__ W,
    const float* __restrict__ bias, float* __restrict__ C,
    unsigned short* __restrict__ Cbf, int M, int N, int K, int act)
{
    __shared__ __align__(16) unsigned short sA[64][40];
    __shared__ __align__(16) unsigned short sB[128][40];

    const int tid  = threadIdx.x;
    const int lane = tid & 31;
    const int wave = tid >> 5;
    const int m0   = blockIdx.y * 64;
    const int n0   = blockIdx.x * 128;
    const int mw   = (wave >> 1) * 32;   // wave m-offset in block tile
    const int nw   = (wave & 1)  * 64;   // wave n-offset in block tile
    const int half = lane >> 4;          // 0: lanes 0-15, 1: lanes 16-31
    const int r    = lane & 15;

    f32x8 acc[2][4];
    const f32x8 vzero = {0.f, 0.f, 0.f, 0.f, 0.f, 0.f, 0.f, 0.f};
#pragma unroll
    for (int i = 0; i < 2; ++i)
#pragma unroll
        for (int j = 0; j < 4; ++j) acc[i][j] = vzero;

    const int srow = tid >> 1;          // 0..63
    const int sseg = (tid & 1) * 16;    // 0 or 16 (bf16 elements)

    for (int k0 = 0; k0 < K; k0 += 32) {
        __syncthreads();
        // Stage bf16 tiles: A 64x32, W 128x32. 32B (2 x b128) per segment.
        {
            const u32x4* pa = (const u32x4*)(A + (size_t)(m0 + srow) * K + k0 + sseg);
            *(u32x4*)&sA[srow][sseg]     = pa[0];
            *(u32x4*)&sA[srow][sseg + 8] = pa[1];
            const u32x4* pw0 = (const u32x4*)(W + (size_t)(n0 + srow) * K + k0 + sseg);
            *(u32x4*)&sB[srow][sseg]     = pw0[0];
            *(u32x4*)&sB[srow][sseg + 8] = pw0[1];
            const u32x4* pw1 = (const u32x4*)(W + (size_t)(n0 + 64 + srow) * K + k0 + sseg);
            *(u32x4*)&sB[64 + srow][sseg]     = pw1[0];
            *(u32x4*)&sB[64 + srow][sseg + 8] = pw1[1];
        }
        if (k0 + 32 < K) {   // hint next K tile (global_prefetch_b8)
            __builtin_prefetch(A + (size_t)(m0 + srow) * K + k0 + 32, 0, 1);
            __builtin_prefetch(W + (size_t)(n0 + srow) * K + k0 + 32, 0, 1);
        }
        __syncthreads();

        // Fragment loads per ISA 7.12.2 (16-bit 16x32 layout):
        // lanes 0-15: elems 0-7 -> K 0-7,  elems 8-15 -> K 16-23
        // lanes16-31: elems 0-7 -> K 8-15, elems 8-15 -> K 24-31
        union Frag { u32x4 q[2]; bf16x16 v; };
        Frag fa[2], fb[4];
#pragma unroll
        for (int i = 0; i < 2; ++i) {
            const unsigned short* p = &sA[mw + i * 16 + r][half * 8];
            fa[i].q[0] = *(const u32x4*)p;
            fa[i].q[1] = *(const u32x4*)(p + 16);
        }
#pragma unroll
        for (int j = 0; j < 4; ++j) {
            const unsigned short* p = &sB[nw + j * 16 + r][half * 8];
            fb[j].q[0] = *(const u32x4*)p;
            fb[j].q[1] = *(const u32x4*)(p + 16);
        }
#pragma unroll
        for (int i = 0; i < 2; ++i)
#pragma unroll
            for (int j = 0; j < 4; ++j)
                acc[i][j] = __builtin_amdgcn_wmma_f32_16x16x32_bf16(
                    false, fa[i].v, false, fb[j].v, (short)0, acc[i][j],
                    false, false);
    }

    // Epilogue. C/D layout: lanes 0-15 -> N=lane, elem e -> M = half*8+e.
#pragma unroll
    for (int i = 0; i < 2; ++i) {
#pragma unroll
        for (int j = 0; j < 4; ++j) {
            const int col = n0 + nw + j * 16 + r;
            const float bb = bias ? bias[col] : 0.0f;
#pragma unroll
            for (int e = 0; e < 8; ++e) {
                const int rowm = m0 + mw + i * 16 + half * 8 + e;
                float v = acc[i][j][e] + bb;
                if (act) v = gelu_tanh(v);
                if (C)   C[(size_t)rowm * N + col] = v;
                if (Cbf) Cbf[(size_t)rowm * N + col] = f2bf(v);
            }
        }
    }
}

// ---------------------------------------------------------------------------
// h[bv*SEQ+l, d] = x_enc[b,l,v]*w_emb[d] + b_emb[d] + w_pos[l,d]  (+bf16 copy)
// ---------------------------------------------------------------------------
__global__ __launch_bounds__(256) void embed_kernel(
    const float* __restrict__ x_enc, const float* __restrict__ w_emb,
    const float* __restrict__ b_emb, const float* __restrict__ w_pos,
    float* __restrict__ h, unsigned short* __restrict__ hb)
{
    const int row = blockIdx.x;            // 0..MROWS-1 = bv*SEQ + l
    const int bv  = row >> 9;
    const int l   = row & (SEQc - 1);
    const int b   = bv / NVARSc;
    const int vv  = bv % NVARSc;
    const float xv = x_enc[((size_t)b * SEQc + l) * NVARSc + vv];
    for (int d = threadIdx.x; d < DMc; d += 256) {
        const float val = xv * w_emb[d] + b_emb[d] + w_pos[(size_t)l * DMc + d];
        h[(size_t)row * DMc + d]  = val;
        hb[(size_t)row * DMc + d] = f2bf(val);
    }
}

// ---------------------------------------------------------------------------
// Sparsity measure: M[bv,h,l] = max_u(q_l . k_{idx(l,u)}) - sum_u(.)/Lk
// ---------------------------------------------------------------------------
__global__ __launch_bounds__(256) void attn_measure_kernel(
    const float* __restrict__ q, const float* __restrict__ k,
    float* __restrict__ Mbuf, int layer)
{
    const int id = blockIdx.x * 256 + threadIdx.x;   // BV*NH*SEQ
    const int l  = id & (SEQc - 1);
    const int bh = id >> 9;
    const int hh = bh & (NHc - 1);
    const int bv = bh >> 3;
    const float* qrow = q + ((size_t)(bv * SEQc + l) * DMc) + hh * DHc;
    float qr[DHc];
#pragma unroll 16
    for (int j = 0; j < DHc; ++j) qr[j] = qrow[j];
    float mx = -3.0e38f, sm = 0.0f;
    for (int u = 0; u < USELc; ++u) {
        const unsigned idx =
            hash_u32((unsigned)l * 977u + (unsigned)u * 1663u +
                     (unsigned)layer * 2753u + 0x9E3779B9u) & (SEQc - 1);
        const float* krow = k + ((size_t)(bv * SEQc + (int)idx) * DMc) + hh * DHc;
        float d = 0.0f;
#pragma unroll 16
        for (int j = 0; j < DHc; ++j) d += qr[j] * krow[j];
        mx = fmaxf(mx, d);
        sm += d;
    }
    Mbuf[id] = mx - sm * (1.0f / (float)SEQc);
}

// ---------------------------------------------------------------------------
// Iterative top-35 (argmax + mask), one workgroup per (bv,h). Deterministic.
// ---------------------------------------------------------------------------
__global__ __launch_bounds__(256) void topk_kernel(
    const float* __restrict__ Mbuf, int* __restrict__ topIdx)
{
    const int bh  = blockIdx.x;            // 0..BV*NH-1
    const int tid = threadIdx.x;
    __shared__ float vals[SEQc];
    __shared__ float rmax[256];
    __shared__ int   rarg[256];
    for (int i = tid; i < SEQc; i += 256) vals[i] = Mbuf[(size_t)bh * SEQc + i];
    __syncthreads();
    for (int it = 0; it < USELc; ++it) {
        float best = -3.0e38f; int bi = SEQc;
        for (int i = tid; i < SEQc; i += 256) {
            float v = vals[i];
            if (v > best || (v == best && i < bi)) { best = v; bi = i; }
        }
        rmax[tid] = best; rarg[tid] = bi;
        __syncthreads();
        for (int s = 128; s > 0; s >>= 1) {
            if (tid < s) {
                if (rmax[tid + s] > rmax[tid] ||
                    (rmax[tid + s] == rmax[tid] && rarg[tid + s] < rarg[tid])) {
                    rmax[tid] = rmax[tid + s]; rarg[tid] = rarg[tid + s];
                }
            }
            __syncthreads();
        }
        if (tid == 0) { topIdx[bh * 64 + it] = rarg[0]; vals[rarg[0]] = -3.0e38f; }
        __syncthreads();
    }
}

// meanv[bv,h,dh] = mean_l v[bv,l,h*64+dh]
__global__ __launch_bounds__(256) void meanv_kernel(
    const float* __restrict__ v, float* __restrict__ meanv)
{
    const int id = blockIdx.x * 256 + threadIdx.x;   // BV*NH*DH = 14336
    if (id >= BVc * NHc * DHc) return;
    const int dh = id & (DHc - 1);
    const int bh = id >> 6;
    const int hh = bh & (NHc - 1);
    const int bv = bh >> 3;
    float s = 0.0f;
    for (int l = 0; l < SEQc; ++l)
        s += v[((size_t)(bv * SEQc + l) * DMc) + hh * DHc + dh];
    meanv[id] = s * (1.0f / (float)SEQc);
}

// ctx[bv,l,:] = broadcast mean_v (all heads)
__global__ __launch_bounds__(256) void ctx_fill_kernel(
    const float* __restrict__ meanv, float* __restrict__ T)
{
    const int row = blockIdx.x;            // MROWS
    const int bv  = row >> 9;
    for (int d = threadIdx.x; d < DMc; d += 256) {
        const int hh = d >> 6, dh = d & (DHc - 1);
        T[(size_t)row * DMc + d] = meanv[((size_t)(bv * NHc + hh)) * DHc + dh];
    }
}

// ---------------------------------------------------------------------------
// Full softmax attention for one selected query: block = (bv,h,ui).
// Overwrites ctx rows at the top-k positions.
// ---------------------------------------------------------------------------
__global__ __launch_bounds__(256) void attn_top_kernel(
    const float* __restrict__ q, const float* __restrict__ k,
    const float* __restrict__ v, const int* __restrict__ topIdx,
    float* __restrict__ T)
{
    const int blk = blockIdx.x;            // BV*NH*USEL
    const int bh  = blk / USELc;
    const int ui  = blk - bh * USELc;
    const int hh  = bh & (NHc - 1);
    const int bv  = bh >> 3;
    const int lq  = topIdx[bh * 64 + ui];
    const int tid = threadIdx.x;

    __shared__ float sq[DHc];
    __shared__ float sp[SEQc];
    __shared__ float red[256];
    __shared__ float s_max, s_sum;

    if (tid < DHc) sq[tid] = q[((size_t)(bv * SEQc + lq) * DMc) + hh * DHc + tid];
    __syncthreads();

    const float scale = 0.125f;            // 1/sqrt(64)
    for (int j = tid; j < SEQc; j += 256) {
        const float* krow = k + ((size_t)(bv * SEQc + j) * DMc) + hh * DHc;
        float d = 0.0f;
#pragma unroll 16
        for (int e = 0; e < DHc; ++e) d += sq[e] * krow[e];
        sp[j] = d * scale;
    }
    __syncthreads();

    float m = fmaxf(sp[tid], sp[tid + 256]);
    red[tid] = m; __syncthreads();
    for (int s = 128; s > 0; s >>= 1) {
        if (tid < s) red[tid] = fmaxf(red[tid], red[tid + s]);
        __syncthreads();
    }
    if (tid == 0) s_max = red[0];
    __syncthreads();

    const float e0 = __expf(sp[tid] - s_max);
    const float e1 = __expf(sp[tid + 256] - s_max);
    sp[tid] = e0; sp[tid + 256] = e1;
    red[tid] = e0 + e1; __syncthreads();
    for (int s = 128; s > 0; s >>= 1) {
        if (tid < s) red[tid] += red[tid + s];
        __syncthreads();
    }
    if (tid == 0) s_sum = red[0];
    __syncthreads();

    const int dh = tid & (DHc - 1), grp = tid >> 6;  // 4 groups of 128 keys
    float acc = 0.0f;
    for (int j = grp * 128; j < grp * 128 + 128; ++j)
        acc += sp[j] * v[((size_t)(bv * SEQc + j) * DMc) + hh * DHc + dh];
    red[tid] = acc; __syncthreads();
    if (grp == 0) {
        const float tot = red[dh] + red[64 + dh] + red[128 + dh] + red[192 + dh];
        T[((size_t)(bv * SEQc + lq) * DMc) + hh * DHc + dh] = tot / s_sum;
    }
}

// ---------------------------------------------------------------------------
// out = LayerNorm(X (+Y)) * g + b over rows of width DM=512. In-place safe.
// Optional bf16 shadow copy for downstream WMMA GEMMs.
// ---------------------------------------------------------------------------
__global__ __launch_bounds__(256) void add_ln_kernel(
    const float* __restrict__ X, const float* __restrict__ Y,
    const float* __restrict__ g, const float* __restrict__ b,
    float* __restrict__ out, unsigned short* __restrict__ outb, int has_res)
{
    const int row = blockIdx.x;
    const int tid = threadIdx.x;
    __shared__ float red[256];
    __shared__ float s_mu, s_rstd;
    const size_t base = (size_t)row * DMc;
    float v0 = X[base + tid];
    float v1 = X[base + tid + 256];
    if (has_res) { v0 += Y[base + tid]; v1 += Y[base + tid + 256]; }
    red[tid] = v0 + v1; __syncthreads();
    for (int s = 128; s > 0; s >>= 1) { if (tid < s) red[tid] += red[tid + s]; __syncthreads(); }
    if (tid == 0) s_mu = red[0] * (1.0f / (float)DMc);
    __syncthreads();
    const float mu = s_mu;
    const float d0 = v0 - mu, d1 = v1 - mu;
    red[tid] = d0 * d0 + d1 * d1; __syncthreads();
    for (int s = 128; s > 0; s >>= 1) { if (tid < s) red[tid] += red[tid + s]; __syncthreads(); }
    if (tid == 0) s_rstd = rsqrtf(red[0] * (1.0f / (float)DMc) + 1e-5f);
    __syncthreads();
    const float rstd = s_rstd;
    const float o0 = d0 * rstd * g[tid]       + b[tid];
    const float o1 = d1 * rstd * g[tid + 256] + b[tid + 256];
    out[base + tid]       = o0;
    out[base + tid + 256] = o1;
    if (outb) {
        outb[base + tid]       = f2bf(o0);
        outb[base + tid + 256] = f2bf(o1);
    }
}

// ---------------------------------------------------------------------------
// Head: out_tmp[bv,p] += sum_k hf[bv,k]*Wh[p,k]  (split-K, Wh read once)
// ---------------------------------------------------------------------------
__global__ __launch_bounds__(256) void head_splitk_kernel(
    const float* __restrict__ hf, const float* __restrict__ Wh,
    float* __restrict__ out_tmp)
{
    const int p     = blockIdx.x;          // 0..PRED-1
    const int chunk = blockIdx.y;          // 0..63, 4096 k each
    const int kbase = chunk * 4096;
    const int tid   = threadIdx.x;
    float w[16];
#pragma unroll
    for (int i = 0; i < 16; ++i)
        w[i] = Wh[(size_t)p * KTOTc + kbase + tid + i * 256];
    __shared__ float red[256];
    for (int bvi = 0; bvi < BVc; ++bvi) {
        const float* hr = hf + (size_t)bvi * KTOTc + kbase;
        float acc = 0.0f;
#pragma unroll
        for (int i = 0; i < 16; ++i) acc += w[i] * hr[tid + i * 256];
        red[tid] = acc; __syncthreads();
        for (int s = 128; s > 0; s >>= 1) { if (tid < s) red[tid] += red[tid + s]; __syncthreads(); }
        if (tid == 0) atomicAdd(&out_tmp[bvi * PREDc + p], red[0]);
        __syncthreads();
    }
}

__global__ __launch_bounds__(256) void head_finalize_kernel(
    const float* __restrict__ out_tmp, const float* __restrict__ bh,
    float* __restrict__ out)
{
    const int id = blockIdx.x * 256 + threadIdx.x;   // (b,p,v) flat
    if (id >= BSZc * PREDc * NVARSc) return;
    const int vv = id % NVARSc;
    const int t  = id / NVARSc;
    const int p  = t % PREDc;
    const int b  = t / PREDc;
    out[id] = out_tmp[(b * NVARSc + vv) * PREDc + p] + bh[p];
}

__global__ __launch_bounds__(256) void zero_kernel(float* __restrict__ p, int n)
{
    const int i = blockIdx.x * 256 + threadIdx.x;
    if (i < n) p[i] = 0.0f;
}

// ---------------------------------------------------------------------------
extern "C" void kernel_launch(void* const* d_in, const int* in_sizes, int n_in,
                              void* d_out, int out_size, void* d_ws, size_t ws_size,
                              hipStream_t stream)
{
    (void)in_sizes; (void)n_in; (void)out_size; (void)ws_size;

    const float* x_enc = (const float*)d_in[0];
    const float* w_emb = (const float*)d_in[1];
    const float* b_emb = (const float*)d_in[2];
    const float* w_pos = (const float*)d_in[3];
    const float* Wq    = (const float*)d_in[4];
    const float* bq    = (const float*)d_in[5];
    const float* Wk    = (const float*)d_in[6];
    const float* bk    = (const float*)d_in[7];
    const float* Wv    = (const float*)d_in[8];
    const float* bvv   = (const float*)d_in[9];
    const float* Wo    = (const float*)d_in[10];
    const float* bo    = (const float*)d_in[11];
    const float* W1    = (const float*)d_in[12];
    const float* b1    = (const float*)d_in[13];
    const float* W2    = (const float*)d_in[14];
    const float* b2    = (const float*)d_in[15];
    const float* g1    = (const float*)d_in[16];
    const float* be1   = (const float*)d_in[17];
    const float* g2    = (const float*)d_in[18];
    const float* be2   = (const float*)d_in[19];
    const float* gF    = (const float*)d_in[20];
    const float* bF    = (const float*)d_in[21];
    const float* Wh    = (const float*)d_in[22];
    const float* bh    = (const float*)d_in[23];

    // Workspace layout (bytes). SB = fp32 row-buffer (14336x512x4 = 28MB).
    //   [0,SB)      h fp32            (persistent)
    //   [SB,2SB)    q fp32 / t1b bf16 (t1b after attention is done with q)
    //   [2SB,4SB)   k,v fp32 / midb bf16 (FFN intermediate, bf16-only)
    //   [4SB,5SB)   t1 fp32 (attention ctx)
    //   [5SB,6SB)   t2 fp32 (projection/FFN2 output)
    //   [6SB,6.5SB) hb bf16 shadow of h
    //   then wb (2MB bf16 weight scratch) and small buffers.
    char* base = (char*)d_ws;
    const size_t SB = (size_t)MROWSc * DMc * 4;
    float* h    = (float*)(base);
    float* q    = (float*)(base + SB);
    float* kbuf = (float*)(base + 2 * SB);
    float* vbuf = (float*)(base + 3 * SB);
    float* t1   = (float*)(base + 4 * SB);
    float* t2   = (float*)(base + 5 * SB);
    unsigned short* t1b  = (unsigned short*)(base + SB);       // overlaps q
    unsigned short* midb = (unsigned short*)(base + 2 * SB);   // overlaps k,v
    unsigned short* hb   = (unsigned short*)(base + 6 * SB);
    unsigned short* wb   = (unsigned short*)(base + 6 * SB + SB / 2);
    float* smallf  = (float*)(base + 6 * SB + SB / 2 + (size_t)DFFc * DMc * 2);
    float* Mbuf    = smallf;                               // BV*NH*SEQ
    int*   topIdx  = (int*)(smallf + BVc * NHc * SEQc);    // BV*NH*64
    float* meanv   = smallf + BVc * NHc * SEQc + BVc * NHc * 64;
    float* out_tmp = meanv + BVc * NHc * DHc;              // BV*PRED

    const int HSn  = MROWSc * DMc;        // 7,340,032
    const int WDMn = DMc * DMc;           // 262,144
    const int WFFn = DFFc * DMc;          // 1,048,576

    zero_kernel<<<(BVc * PREDc + 255) / 256, 256, 0, stream>>>(out_tmp, BVc * PREDc);
    embed_kernel<<<MROWSc, 256, 0, stream>>>(x_enc, w_emb, b_emb, w_pos, h, hb);

    const dim3 gqkv(DMc / 128, MROWSc / 64);    // (4, 224)
    const dim3 gffn1(DFFc / 128, MROWSc / 64);  // (16, 224)
    const int cvW  = (WDMn / 4 + 255) / 256;
    const int cvF  = (WFFn / 4 + 255) / 256;
    const int cvH  = (HSn / 4 + 255) / 256;

    for (int l = 0; l < ELc; ++l) {
        const float* Wq_l = Wq + (size_t)l * WDMn;  const float* bq_l = bq + l * DMc;
        const float* Wk_l = Wk + (size_t)l * WDMn;  const float* bk_l = bk + l * DMc;
        const float* Wv_l = Wv + (size_t)l * WDMn;  const float* bv_l = bvv + l * DMc;
        const float* Wo_l = Wo + (size_t)l * WDMn;  const float* bo_l = bo + l * DMc;
        const float* W1_l = W1 + (size_t)l * WFFn;  const float* b1_l = b1 + l * DFFc;
        const float* W2_l = W2 + (size_t)l * WFFn;  const float* b2_l = b2 + l * DMc;
        const float* g1_l = g1 + l * DMc;  const float* be1_l = be1 + l * DMc;
        const float* g2_l = g2 + l * DMc;  const float* be2_l = be2 + l * DMc;

        // Q, K, V projections (bf16 WMMA, fp32 out for scalar attention)
        convert_bf16_kernel<<<cvW, 256, 0, stream>>>(Wq_l, wb, WDMn);
        gemm_bf16_wmma<<<gqkv, 128, 0, stream>>>(hb, wb, bq_l, q,    nullptr, MROWSc, DMc, DMc, 0);
        convert_bf16_kernel<<<cvW, 256, 0, stream>>>(Wk_l, wb, WDMn);
        gemm_bf16_wmma<<<gqkv, 128, 0, stream>>>(hb, wb, bk_l, kbuf, nullptr, MROWSc, DMc, DMc, 0);
        convert_bf16_kernel<<<cvW, 256, 0, stream>>>(Wv_l, wb, WDMn);
        gemm_bf16_wmma<<<gqkv, 128, 0, stream>>>(hb, wb, bv_l, vbuf, nullptr, MROWSc, DMc, DMc, 0);

        // ProbSparse attention (scalar; <1% of FLOPs)
        attn_measure_kernel<<<(BVc * NHc * SEQc) / 256, 256, 0, stream>>>(q, kbuf, Mbuf, l);
        topk_kernel<<<BVc * NHc, 256, 0, stream>>>(Mbuf, topIdx);
        meanv_kernel<<<(BVc * NHc * DHc + 255) / 256, 256, 0, stream>>>(vbuf, meanv);
        ctx_fill_kernel<<<MROWSc, 256, 0, stream>>>(meanv, t1);
        attn_top_kernel<<<BVc * NHc * USELc, 256, 0, stream>>>(q, kbuf, vbuf, topIdx, t1);

        // O projection (q is dead -> its region hosts t1b)
        convert_bf16_kernel<<<cvH, 256, 0, stream>>>(t1, t1b, HSn);
        convert_bf16_kernel<<<cvW, 256, 0, stream>>>(Wo_l, wb, WDMn);
        gemm_bf16_wmma<<<gqkv, 128, 0, stream>>>(t1b, wb, bo_l, t2, nullptr, MROWSc, DMc, DMc, 0);
        add_ln_kernel<<<MROWSc, 256, 0, stream>>>(h, t2, g1_l, be1_l, h, hb, 1);

        // FFN: intermediate produced directly as bf16 (GELU in epilogue)
        convert_bf16_kernel<<<cvF, 256, 0, stream>>>(W1_l, wb, WFFn);
        gemm_bf16_wmma<<<gffn1, 128, 0, stream>>>(hb, wb, b1_l, nullptr, midb, MROWSc, DFFc, DMc, 1);
        convert_bf16_kernel<<<cvF, 256, 0, stream>>>(W2_l, wb, WFFn);
        gemm_bf16_wmma<<<gqkv, 128, 0, stream>>>(midb, wb, b2_l, t2, nullptr, MROWSc, DMc, DFFc, 0);
        add_ln_kernel<<<MROWSc, 256, 0, stream>>>(h, t2, g2_l, be2_l, h, hb, 1);
    }

    // Final LN into q (free after the loop), then head projection.
    add_ln_kernel<<<MROWSc, 256, 0, stream>>>(h, h, gF, bF, q, nullptr, 0);
    const dim3 ghead(PREDc, 64);
    head_splitk_kernel<<<ghead, 256, 0, stream>>>(q, Wh, out_tmp);
    head_finalize_kernel<<<(BSZc * PREDc * NVARSc + 255) / 256, 256, 0, stream>>>(
        out_tmp, bh, (float*)d_out);
}
